// Decoder_25975962206235
// MI455X (gfx1250) — matrix operand order repaired and checked by reading, not compiled, not run on hardware
//
#include <hip/hip_runtime.h>
#include <hip/hip_bf16.h>
#include <cstdint>
#include <cstddef>

// ---------------- problem constants ----------------
#define BB     32
#define TENC   512
#define TDEC   512
#define DD     512
#define MELN   80
#define KCONV  31

#define KATT   1120   // 592 + 512 padded to 32-multiple (1104 -> 1120)
#define KDEC   1536   // 1024 + 512
#define KLOC   64     // 2*31 padded to 64

typedef __attribute__((ext_vector_type(16))) _Float16 v16h;
typedef __attribute__((ext_vector_type(8)))  _Float16 v8h;
typedef __attribute__((ext_vector_type(8)))  float    v8f;

// ---------------- WMMA tile loaders (ISA 7.12.2 layouts) ----------------
// A 16x32 f16: lane<16 holds K {0..7,16..23}; lane>=16 holds K {8..15,24..31}
__device__ __forceinline__ v16h load_a16(const _Float16* row_k0, int lane) {
  int khi = (lane >> 4) & 1;                       // 0 or 1
  v8h lo = *(const v8h*)(row_k0 + khi * 8);        // K = khi*8 .. +7
  v8h hi = *(const v8h*)(row_k0 + 16 + khi * 8);   // K = 16+khi*8 .. +7
  v16h r;
#pragma unroll
  for (int i = 0; i < 8; ++i) { r[i] = lo[i]; r[i + 8] = hi[i]; }
  return r;
}
// B 32x16 f16: same K order as A: lane<16 holds K {0..7,16..23};
// lane>=16 holds K {8..15,24..31}.
__device__ __forceinline__ v16h load_b16(const _Float16* row_k0, int lane) {
  int khi = (lane >> 4) & 1;
  v8h lo = *(const v8h*)(row_k0 + khi * 8);
  v8h hi = *(const v8h*)(row_k0 + 16 + khi * 8);
  v16h r;
#pragma unroll
  for (int i = 0; i < 8; ++i) { r[i] = lo[i]; r[i + 8] = hi[i]; }
  return r;
}

// ---------------- generic WMMA GEMM: C[M,N] = A[M,K] * W[N,K]^T (+bias[N]) ----
// One wave computes a 16x64 strip (4 accumulators, A fragment reused 4x).
// K multiple of 32, M multiple of 16, N multiple of 64.
__global__ void gemm_wmma_f16(const _Float16* __restrict__ A,
                              const _Float16* __restrict__ W,
                              const float* __restrict__ bias,
                              float* __restrict__ C,
                              int M, int N, int K) {
  int wave = (blockIdx.x * blockDim.x + threadIdx.x) >> 5;
  int lane = threadIdx.x & 31;
  int tilesN = N >> 6;                       // 64-wide N strips
  int total = (M >> 4) * tilesN;
  if (wave >= total) return;                 // wave-uniform: EXEC stays all-ones
  int tm = wave / tilesN;
  int tn = wave - tm * tilesN;

  const _Float16* arow = A + (size_t)(tm * 16 + (lane & 15)) * K;
  const _Float16* brow = W + (size_t)(tn * 64 + (lane & 15)) * K;
  const size_t bstride = (size_t)16 * K;     // 16 N-rows per sub-tile

  v8f acc0 = {}, acc1 = {}, acc2 = {}, acc3 = {};
  for (int k0 = 0; k0 < K; k0 += 32) {
    // speculative prefetch of next K block (global_prefetch_b8); OOB is dropped
    __builtin_prefetch(arow + k0 + 32, 0, 0);
    __builtin_prefetch(brow + k0 + 32, 0, 0);
    v16h a  = load_a16(arow + k0, lane);
    v16h b0 = load_b16(brow + k0, lane);
    v16h b1 = load_b16(brow + bstride + k0, lane);
    v16h b2 = load_b16(brow + 2 * bstride + k0, lane);
    v16h b3 = load_b16(brow + 3 * bstride + k0, lane);
    acc0 = __builtin_amdgcn_wmma_f32_16x16x32_f16(false, a, false, b0, (short)0, acc0, false, false);
    acc1 = __builtin_amdgcn_wmma_f32_16x16x32_f16(false, a, false, b1, (short)0, acc1, false, false);
    acc2 = __builtin_amdgcn_wmma_f32_16x16x32_f16(false, a, false, b2, (short)0, acc2, false, false);
    acc3 = __builtin_amdgcn_wmma_f32_16x16x32_f16(false, a, false, b3, (short)0, acc3, false, false);
  }

  int ncol = lane & 15;
  int m0 = tm * 16 + ((lane >> 4) ? 8 : 0);
  v8f accs[4] = {acc0, acc1, acc2, acc3};
#pragma unroll
  for (int j = 0; j < 4; ++j) {
    int n = tn * 64 + j * 16 + ncol;
    float bv = bias ? bias[n] : 0.0f;
#pragma unroll
    for (int r = 0; r < 8; ++r)
      C[(size_t)(m0 + r) * N + n] = accs[j][r] + bv;
  }
}

// ---------------- one-time packing / casting ----------------
__global__ void pack_att_w(const float* __restrict__ Wih, const float* __restrict__ Whh,
                           _Float16* __restrict__ out) {
  int idx = blockIdx.x * blockDim.x + threadIdx.x;   // 2048 * KATT
  if (idx >= 4 * DD * KATT) return;
  int j = idx / KATT, k = idx - j * KATT;
  float v = 0.0f;
  if (k < DD + MELN)            v = Wih[(size_t)j * (DD + MELN) + k];
  else if (k < DD + MELN + DD)  v = Whh[(size_t)j * DD + (k - (DD + MELN))];
  out[idx] = (_Float16)v;
}

__global__ void pack_dec_w(const float* __restrict__ Wih, const float* __restrict__ Whh,
                           _Float16* __restrict__ out) {
  int idx = blockIdx.x * blockDim.x + threadIdx.x;   // 2048 * KDEC
  if (idx >= 4 * DD * KDEC) return;
  int j = idx / KDEC, k = idx - j * KDEC;
  float v = (k < 2 * DD) ? Wih[(size_t)j * (2 * DD) + k]
                         : Whh[(size_t)j * DD + (k - 2 * DD)];
  out[idx] = (_Float16)v;
}

__global__ void pack_loc_w(const float* __restrict__ Wloc, _Float16* __restrict__ out) {
  int idx = blockIdx.x * blockDim.x + threadIdx.x;   // 512 * 64
  if (idx >= DD * KLOC) return;
  int d = idx >> 6, j = idx & 63;
  int c = j >> 5, k = j & 31;
  float v = (k < KCONV) ? Wloc[((size_t)d * 2 + c) * KCONV + k] : 0.0f;
  out[idx] = (_Float16)v;
}

__global__ void cast_f32_f16(const float* __restrict__ in, _Float16* __restrict__ out, int n) {
  int idx = blockIdx.x * blockDim.x + threadIdx.x;
  if (idx < n) out[idx] = (_Float16)in[idx];
}

__global__ void zero_f32(float* __restrict__ p, int n) {
  int idx = blockIdx.x * blockDim.x + threadIdx.x;
  if (idx < n) p[idx] = 0.0f;
}

// ---------------- per-step kernels ----------------
__global__ void build_xatt(const float* __restrict__ mel_in, int t,
                           const float* __restrict__ ctx, const float* __restrict__ h_att,
                           _Float16* __restrict__ x) {
  int idx = blockIdx.x * blockDim.x + threadIdx.x;   // 32 * KATT
  if (idx >= BB * KATT) return;
  int b = idx / KATT, k = idx - b * KATT;
  float v = 0.0f;
  if (k < MELN)                 v = mel_in[((size_t)b * TDEC + t) * MELN + k];
  else if (k < MELN + DD)       v = ctx[b * DD + (k - MELN)];
  else if (k < MELN + 2 * DD)   v = h_att[b * DD + (k - MELN - DD)];
  x[idx] = (_Float16)v;
}

__global__ void build_xdec(const float* __restrict__ h_att, const float* __restrict__ ctx,
                           const float* __restrict__ h_dec, _Float16* __restrict__ x) {
  int idx = blockIdx.x * blockDim.x + threadIdx.x;   // 32 * KDEC
  if (idx >= BB * KDEC) return;
  int b = idx / KDEC, k = idx - b * KDEC;
  float v;
  if (k < DD)            v = h_att[b * DD + k];
  else if (k < 2 * DD)   v = ctx[b * DD + (k - DD)];
  else                   v = h_dec[b * DD + (k - 2 * DD)];
  x[idx] = (_Float16)v;
}

__device__ __forceinline__ float sigm(float x) { return 1.0f / (1.0f + expf(-x)); }

__global__ void lstm_point(const float* __restrict__ gates,   // [B,4D], bias already added
                           float* __restrict__ h, float* __restrict__ c,
                           _Float16* __restrict__ h16) {
  int idx = blockIdx.x * blockDim.x + threadIdx.x;   // 32*512
  if (idx >= BB * DD) return;
  int b = idx >> 9, d = idx & (DD - 1);
  const float* g = gates + (size_t)b * 4 * DD;
  float i_ = sigm(g[d]);
  float f_ = sigm(g[DD + d]);
  float gg = tanhf(g[2 * DD + d]);
  float o_ = sigm(g[3 * DD + d]);
  float cn = f_ * c[idx] + i_ * gg;
  float hn = o_ * tanhf(cn);
  c[idx] = cn;
  h[idx] = hn;
  if (h16) h16[idx] = (_Float16)hn;
}

// im2col for location conv: X[(b*Tenc+te)][c*32+k] = {a|a_cum}[b][te+k-15]
__global__ void build_im2col(const float* __restrict__ a, const float* __restrict__ a_cum,
                             _Float16* __restrict__ X) {
  int idx = blockIdx.x * blockDim.x + threadIdx.x;   // 16384 * 64
  if (idx >= BB * TENC * KLOC) return;
  int row = idx >> 6, j = idx & 63;
  int b = row >> 9, te = row & (TENC - 1);
  int c = j >> 5, k = j & 31;
  float v = 0.0f;
  if (k < KCONV) {
    int src = te + k - (KCONV / 2);
    if (src >= 0 && src < TENC) v = (c == 0 ? a : a_cum)[b * TENC + src];
  }
  X[idx] = (_Float16)v;
}

// e[b,te] = sum_d v[d] * tanh(q[b,d] + m_proj[b,te,d] + loc[b,te,d]); wave per row
__global__ void energy_kernel(const float* __restrict__ q, const float* __restrict__ m_proj,
                              const float* __restrict__ loc, const float* __restrict__ vvec,
                              float* __restrict__ e) {
  int wave = (blockIdx.x * blockDim.x + threadIdx.x) >> 5;
  int lane = threadIdx.x & 31;
  if (wave >= BB * TENC) return;
  int b = wave >> 9;
  const float* qb = q + (size_t)b * DD;
  const float* mp = m_proj + (size_t)wave * DD;
  const float* lp = loc + (size_t)wave * DD;
  float acc = 0.0f;
  for (int d = lane; d < DD; d += 32)
    acc += vvec[d] * tanhf(qb[d] + mp[d] + lp[d]);
#pragma unroll
  for (int off = 16; off > 0; off >>= 1) acc += __shfl_xor(acc, off, 32);
  if (lane == 0) e[wave] = acc;
}

// per-batch softmax over Tenc, context = a_new @ enc[b], a_cum += a_new
__global__ void softmax_ctx_kernel(const float* __restrict__ e, const float* __restrict__ enc,
                                   float* __restrict__ a, float* __restrict__ a_cum,
                                   float* __restrict__ ctx) {
  __shared__ float red[256];
  __shared__ float an[TENC];
  int b = blockIdx.x, tid = threadIdx.x;
  const float* eb = e + (size_t)b * TENC;
  float m = -1e30f;
  for (int t = tid; t < TENC; t += 256) m = fmaxf(m, eb[t]);
  red[tid] = m; __syncthreads();
  for (int s = 128; s > 0; s >>= 1) { if (tid < s) red[tid] = fmaxf(red[tid], red[tid + s]); __syncthreads(); }
  m = red[0]; __syncthreads();
  float sum = 0.0f;
  for (int t = tid; t < TENC; t += 256) { float ex = expf(eb[t] - m); an[t] = ex; sum += ex; }
  red[tid] = sum; __syncthreads();
  for (int s = 128; s > 0; s >>= 1) { if (tid < s) red[tid] += red[tid + s]; __syncthreads(); }
  float inv = 1.0f / red[0]; __syncthreads();
  for (int t = tid; t < TENC; t += 256) {
    float av = an[t] * inv;
    an[t] = av;
    a[b * TENC + t] = av;
    a_cum[b * TENC + t] += av;
  }
  __syncthreads();
  for (int d = tid; d < DD; d += 256) {
    float acc = 0.0f;
    const float* encb = enc + ((size_t)b * TENC) * DD + d;
    for (int t = 0; t < TENC; ++t) acc += an[t] * encb[(size_t)t * DD];
    ctx[b * DD + d] = acc;
  }
}

__global__ void melstop_kernel(const float* __restrict__ h_dec,
                               const float* __restrict__ W_mel, const float* __restrict__ b_mel,
                               const float* __restrict__ W_stop, const float* __restrict__ b_stop,
                               float* __restrict__ out, int t) {
  int idx = blockIdx.x * blockDim.x + threadIdx.x;
  if (idx < BB * MELN) {
    int b = idx / MELN, m = idx - b * MELN;
    const float* hb = h_dec + (size_t)b * DD;
    const float* wm = W_mel + (size_t)m * DD;
    float acc = b_mel[m];
    for (int d = 0; d < DD; ++d) acc += hb[d] * wm[d];
    out[((size_t)b * TDEC + t) * MELN + m] = acc;
  } else if (idx < BB * MELN + BB) {
    int b = idx - BB * MELN;
    const float* hb = h_dec + (size_t)b * DD;
    float acc = b_stop[0];
    for (int d = 0; d < DD; ++d) acc += hb[d] * W_stop[d];
    out[(size_t)BB * TDEC * MELN + (size_t)b * TDEC + t] = acc;
  }
}

// ---------------- host orchestration ----------------
static inline int cdiv(long long a, long long b) { return (int)((a + b - 1) / b); }

extern "C" void kernel_launch(void* const* d_in, const int* in_sizes, int n_in,
                              void* d_out, int out_size, void* d_ws, size_t ws_size,
                              hipStream_t stream) {
  (void)in_sizes; (void)n_in; (void)out_size; (void)ws_size;
  const float* enc      = (const float*)d_in[0];
  const float* mel_in   = (const float*)d_in[1];
  const float* W_ih_att = (const float*)d_in[2];
  const float* W_hh_att = (const float*)d_in[3];
  const float* b_att    = (const float*)d_in[4];
  const float* W_ih_dec = (const float*)d_in[5];
  const float* W_hh_dec = (const float*)d_in[6];
  const float* b_dec    = (const float*)d_in[7];
  const float* Wq       = (const float*)d_in[8];
  const float* Wm       = (const float*)d_in[9];
  const float* W_loc    = (const float*)d_in[10];
  const float* vvec     = (const float*)d_in[11];
  const float* W_mel    = (const float*)d_in[12];
  const float* b_mel    = (const float*)d_in[13];
  const float* W_stop   = (const float*)d_in[14];
  const float* b_stop   = (const float*)d_in[15];
  float* out = (float*)d_out;

  // workspace carve-up (256B aligned)
  char* w = (char*)d_ws;
  auto alloc = [&](size_t bytes) { char* p = w; w += (bytes + 255) & ~(size_t)255; return p; };
  _Float16* Watt16  = (_Float16*)alloc((size_t)4 * DD * KATT * 2);
  _Float16* Wdec16  = (_Float16*)alloc((size_t)4 * DD * KDEC * 2);
  _Float16* Wq16    = (_Float16*)alloc((size_t)DD * DD * 2);
  _Float16* Wm16    = (_Float16*)alloc((size_t)DD * DD * 2);
  _Float16* Wl16    = (_Float16*)alloc((size_t)DD * KLOC * 2);
  _Float16* enc16   = (_Float16*)alloc((size_t)BB * TENC * DD * 2);
  float*    m_proj  = (float*)alloc((size_t)BB * TENC * DD * 4);
  float*    loc     = (float*)alloc((size_t)BB * TENC * DD * 4);
  _Float16* Xloc    = (_Float16*)alloc((size_t)BB * TENC * KLOC * 2);
  float*    gates   = (float*)alloc((size_t)BB * 4 * DD * 4);
  float*    q       = (float*)alloc((size_t)BB * DD * 4);
  _Float16* xatt16  = (_Float16*)alloc((size_t)BB * KATT * 2);
  _Float16* xdec16  = (_Float16*)alloc((size_t)BB * KDEC * 2);
  _Float16* hatt16  = (_Float16*)alloc((size_t)BB * DD * 2);
  float*    evec    = (float*)alloc((size_t)BB * TENC * 4);
  float*    states  = (float*)alloc((size_t)7 * BB * DD * 4);  // note: BB*TENC == BB*DD here
  float* ctx   = states + 0 * BB * DD;
  float* aatt  = states + 1 * BB * DD;   // [B, Tenc]
  float* acum  = states + 2 * BB * DD;   // [B, Tenc]
  float* h_att = states + 3 * BB * DD;
  float* c_att = states + 4 * BB * DD;
  float* h_dec = states + 5 * BB * DD;
  float* c_dec = states + 6 * BB * DD;

  const int TPB = 256;

  // ---- one-time packing / casting / m_proj ----
  pack_att_w<<<cdiv((long long)4 * DD * KATT, TPB), TPB, 0, stream>>>(W_ih_att, W_hh_att, Watt16);
  pack_dec_w<<<cdiv((long long)4 * DD * KDEC, TPB), TPB, 0, stream>>>(W_ih_dec, W_hh_dec, Wdec16);
  pack_loc_w<<<cdiv(DD * KLOC, TPB), TPB, 0, stream>>>(W_loc, Wl16);
  cast_f32_f16<<<cdiv(DD * DD, TPB), TPB, 0, stream>>>(Wq, Wq16, DD * DD);
  cast_f32_f16<<<cdiv(DD * DD, TPB), TPB, 0, stream>>>(Wm, Wm16, DD * DD);
  cast_f32_f16<<<cdiv((long long)BB * TENC * DD, TPB), TPB, 0, stream>>>(enc, enc16, BB * TENC * DD);
  zero_f32<<<cdiv(7 * BB * DD, TPB), TPB, 0, stream>>>(states, 7 * BB * DD);

  // m_proj = enc @ Wm^T : M=16384, N=512, K=512 -> 8192 waves (16x64 strips)
  {
    int waves = (BB * TENC / 16) * (DD / 64);
    gemm_wmma_f16<<<cdiv((long long)waves * 32, TPB), TPB, 0, stream>>>(
        enc16, Wm16, nullptr, m_proj, BB * TENC, DD, DD);
  }

  // ---- sequential decode ----
  for (int t = 0; t < TDEC; ++t) {
    // attention LSTM
    build_xatt<<<cdiv(BB * KATT, TPB), TPB, 0, stream>>>(mel_in, t, ctx, h_att, xatt16);
    {
      int waves = (BB / 16) * (4 * DD / 64);   // 64
      gemm_wmma_f16<<<cdiv((long long)waves * 32, TPB), TPB, 0, stream>>>(
          xatt16, Watt16, b_att, gates, BB, 4 * DD, KATT);
    }
    lstm_point<<<cdiv(BB * DD, TPB), TPB, 0, stream>>>(gates, h_att, c_att, hatt16);

    // q = h_att @ Wq^T
    {
      int waves = (BB / 16) * (DD / 64);       // 16
      gemm_wmma_f16<<<cdiv((long long)waves * 32, TPB), TPB, 0, stream>>>(
          hatt16, Wq16, nullptr, q, BB, DD, DD);
    }

    // location features: im2col + WMMA GEMM (dominant FLOPs of the step)
    build_im2col<<<cdiv((long long)BB * TENC * KLOC, TPB), TPB, 0, stream>>>(aatt, acum, Xloc);
    {
      int waves = (BB * TENC / 16) * (DD / 64);  // 8192
      gemm_wmma_f16<<<cdiv((long long)waves * 32, TPB), TPB, 0, stream>>>(
          Xloc, Wl16, nullptr, loc, BB * TENC, DD, KLOC);
    }

    // energies, softmax, context
    energy_kernel<<<cdiv((long long)BB * TENC * 32, TPB), TPB, 0, stream>>>(q, m_proj, loc, vvec, evec);
    softmax_ctx_kernel<<<BB, TPB, 0, stream>>>(evec, enc, aatt, acum, ctx);

    // decoder LSTM
    build_xdec<<<cdiv(BB * KDEC, TPB), TPB, 0, stream>>>(h_att, ctx, h_dec, xdec16);
    {
      int waves = (BB / 16) * (4 * DD / 64);   // 64
      gemm_wmma_f16<<<cdiv((long long)waves * 32, TPB), TPB, 0, stream>>>(
          xdec16, Wdec16, b_dec, gates, BB, 4 * DD, KDEC);
    }
    lstm_point<<<cdiv(BB * DD, TPB), TPB, 0, stream>>>(gates, h_dec, c_dec, nullptr);

    // outputs
    melstop_kernel<<<cdiv(BB * MELN + BB, TPB), TPB, 0, stream>>>(
        h_dec, W_mel, b_mel, W_stop, b_stop, out, t);
  }
}
